// lstm_71270687310298
// MI455X (gfx1250) — compile-verified, matrix operand order checked by reading, not run on hardware
//
#include <hip/hip_runtime.h>

// ---------- types ----------
typedef __attribute__((ext_vector_type(16))) __bf16 v16bf;
typedef __attribute__((ext_vector_type(8)))  float  v8f;

struct alignas(16) U128 { unsigned int a, b, c, d; };          // 16B LDS/global chunk
struct alignas(16) F128 { float x, y, z, w; };                 // 16B float chunk
struct alignas(8)  US4  { unsigned short x, y, z, w; };        // 8B bf16x4 chunk

union Frag { v16bf v; U128 q[2]; };                            // 32B WMMA operand

// ---------- problem sizes ----------
#define TB 512
#define TI 2048
#define TH 2048
#define TK 4096   // I + H (concat)

// ---------- GEMM tiling ----------
#define BM 128
#define BN 128
#define BK 32
#define LDH 40    // padded LDS row stride in halves (80B, keeps 16B alignment)

static __device__ __forceinline__ unsigned short to_bf16(float f) {
  unsigned int u = __float_as_uint(f);
  u += 0x7FFFu + ((u >> 16) & 1u);   // round-to-nearest-even
  return (unsigned short)(u >> 16);
}

// ============================================================
// Pass 0: pack X = [inputs | prevstate] into bf16 [512 x 4096]
// ============================================================
__global__ void __launch_bounds__(256)
lstm_pack_x(const float* __restrict__ inp, const float* __restrict__ st0,
            unsigned short* __restrict__ Xbf) {
  int t = blockIdx.x * 256 + threadIdx.x;        // 0 .. 512*4096/4 - 1
  int idx = t << 2;
  int m = idx >> 12;                              // / 4096
  int k = idx & (TK - 1);
  const float* src = (k < TI) ? (inp + (size_t)m * TI + k)
                              : (st0 + (size_t)m * TH + (k - TI));
  F128 v = *reinterpret_cast<const F128*>(src);
  US4 o;
  o.x = to_bf16(v.x); o.y = to_bf16(v.y); o.z = to_bf16(v.z); o.w = to_bf16(v.w);
  *reinterpret_cast<US4*>(Xbf + (size_t)m * TK + k) = o;
}

// ============================================================
// Stage one K-block: A (bf16 copy) and B (f32 -> bf16, transposed)
// ============================================================
static __device__ __forceinline__ void
stage_tiles(unsigned short* __restrict__ Adst, unsigned short* __restrict__ Bdst,
            const unsigned short* __restrict__ Xbf,
            const float* __restrict__ Wp, const float* __restrict__ Up,
            int m0, int n0, int k0, int tid, int n4, int kr) {
  // A tile: 128 rows x 32 halves, 16B chunks, 2 per thread
  #pragma unroll
  for (int c = 0; c < 2; ++c) {
    int ch  = tid + c * 256;            // 0..511
    int row = ch >> 2;
    int ko  = (ch & 3) * 8;
    U128 v = *reinterpret_cast<const U128*>(
        Xbf + (size_t)(m0 + row) * TK + k0 + ko);
    *reinterpret_cast<U128*>(&Adst[row * LDH + ko]) = v;
  }
  // B tile: 32 k-rows x 128 n f32 -> bf16, stored [n][k]
  const float* Wsrc = (k0 < TI) ? (Wp + (size_t)k0 * TH)
                                : (Up + (size_t)(k0 - TI) * TH);
  float vt[4][4];
  #pragma unroll
  for (int r = 0; r < 4; ++r) {
    F128 t = *reinterpret_cast<const F128*>(
        Wsrc + (size_t)(kr + r) * TH + n0 + n4);
    vt[r][0] = t.x; vt[r][1] = t.y; vt[r][2] = t.z; vt[r][3] = t.w;
  }
  #pragma unroll
  for (int j = 0; j < 4; ++j) {
    US4 p;
    p.x = to_bf16(vt[0][j]);
    p.y = to_bf16(vt[1][j]);
    p.z = to_bf16(vt[2][j]);
    p.w = to_bf16(vt[3][j]);
    *reinterpret_cast<US4*>(&Bdst[(n4 + j) * LDH + kr]) = p;
  }
}

// ============================================================
// Pass 1: Z[g] = X @ [Wg ; Ug]   (bf16 WMMA, f32 accumulate)
// grid = (TH/BN, TB/BM, 4 gates), block = 256 (8 waves)
// Double-buffered LDS: one barrier per K-block.
// ============================================================
__global__ void __launch_bounds__(256)
lstm_gemm4(const unsigned short* __restrict__ Xbf,
           const float* __restrict__ Wi, const float* __restrict__ Ui,
           const float* __restrict__ Wf, const float* __restrict__ Uf,
           const float* __restrict__ Wg, const float* __restrict__ Ug,
           const float* __restrict__ Wc, const float* __restrict__ Uc,
           float* __restrict__ Z) {
  __shared__ unsigned short Asm[2][BM * LDH];   // [buf][m][k], 2 x 10KB
  __shared__ unsigned short Bsm[2][BN * LDH];   // [buf][n][k], 2 x 10KB

  const int g = blockIdx.z;
  const float* Wp; const float* Up;
  switch (g) {
    case 0:  Wp = Wi; Up = Ui; break;
    case 1:  Wp = Wf; Up = Uf; break;
    case 2:  Wp = Wg; Up = Ug; break;
    default: Wp = Wc; Up = Uc; break;
  }

  const int tid  = threadIdx.x;
  const int lane = tid & 31;
  const int wave = tid >> 5;
  const int wr = wave >> 2;          // 0..1  (64-row band)
  const int wc = wave & 3;           // 0..3  (32-col band)
  const int lo = lane & 15;          // M/N within 16x16 tile
  const int hi = lane >> 4;          // selects K sub-range per ISA layout

  const int m0 = blockIdx.y * BM;
  const int n0 = blockIdx.x * BN;

  // B-tile load assignment: each thread covers 4 k-rows x 4 n-cols
  const int n4 = (tid & 31) * 4;     // 0..124
  const int kr = (tid >> 5) * 4;     // 0..28

  v8f acc[4][2] = {};

  // prologue: fill buffer 0
  stage_tiles(Asm[0], Bsm[0], Xbf, Wp, Up, m0, n0, 0, tid, n4, kr);
  __syncthreads();

  int buf = 0;
  for (int k0 = 0; k0 < TK; k0 += BK) {
    const int nxt = buf ^ 1;
    // stage next K-block into the other buffer (overlaps with WMMAs below)
    if (k0 + BK < TK)
      stage_tiles(Asm[nxt], Bsm[nxt], Xbf, Wp, Up, m0, n0, k0 + BK, tid, n4, kr);
    // prefetch two K-blocks ahead toward the WGP
    if (k0 + 2 * BK < TK) {
      const int kp = k0 + 2 * BK;
      const float* WsrcP = (kp < TI) ? (Wp + (size_t)kp * TH)
                                     : (Up + (size_t)(kp - TI) * TH);
      __builtin_prefetch(WsrcP + (size_t)kr * TH + n0 + n4, 0, 3);
      __builtin_prefetch(Xbf + (size_t)(m0 + (tid >> 1)) * TK + kp, 0, 3);
    }

    // ---- fragments per ISA 16-bit A/B layout; 16B ds_load chunks ----
    const unsigned short* Acur = Asm[buf];
    const unsigned short* Bcur = Bsm[buf];
    Frag a[4], b[2];
    #pragma unroll
    for (int i = 0; i < 4; ++i) {
      const unsigned short* ap = &Acur[(wr * 64 + i * 16 + lo) * LDH + hi * 8];
      a[i].q[0] = *reinterpret_cast<const U128*>(ap);       // K = hi*8 .. +7
      a[i].q[1] = *reinterpret_cast<const U128*>(ap + 16);  // K = 16+hi*8 .. +7
    }
    #pragma unroll
    for (int j = 0; j < 2; ++j) {
      const unsigned short* bp = &Bcur[(wc * 32 + j * 16 + lo) * LDH + hi * 8];
      b[j].q[0] = *reinterpret_cast<const U128*>(bp);
      b[j].q[1] = *reinterpret_cast<const U128*>(bp + 16);
    }
    #pragma unroll
    for (int i = 0; i < 4; ++i)
      #pragma unroll
      for (int j = 0; j < 2; ++j)
        acc[i][j] = __builtin_amdgcn_wmma_f32_16x16x32_bf16(
            false, a[i].v, false, b[j].v, (short)0, acc[i][j], false, false);

    __syncthreads();   // next buffer staged AND current buffer reads done
    buf = nxt;
  }

  // ---- write preactivations: Z[g][m][n], C/D layout: VGPR r -> M = hi*8+r ----
  float* Zg = Z + (size_t)g * ((size_t)TB * TH);
  #pragma unroll
  for (int i = 0; i < 4; ++i) {
    int mb = m0 + wr * 64 + i * 16 + hi * 8;
    #pragma unroll
    for (int j = 0; j < 2; ++j) {
      int n = n0 + wc * 32 + j * 16 + lo;
      #pragma unroll
      for (int r = 0; r < 8; ++r)
        Zg[(size_t)(mb + r) * TH + n] = acc[i][j][r];
    }
  }
}

// ============================================================
// Pass 2: gates -> c, state.  out = [c | state | c]
// ============================================================
__global__ void __launch_bounds__(256)
lstm_eltwise(const float* __restrict__ Z,
             const float* __restrict__ bi, const float* __restrict__ bfv,
             const float* __restrict__ bg, const float* __restrict__ bc,
             const float* __restrict__ prevout, float* __restrict__ out) {
  const size_t BH = (size_t)TB * TH;
  size_t idx = ((size_t)blockIdx.x * 256 + threadIdx.x) << 2;
  int n = (int)(idx & (TH - 1));

  F128 zi = *reinterpret_cast<const F128*>(Z + idx);
  F128 zf = *reinterpret_cast<const F128*>(Z + BH + idx);
  F128 zg = *reinterpret_cast<const F128*>(Z + 2 * BH + idx);
  F128 zc = *reinterpret_cast<const F128*>(Z + 3 * BH + idx);
  F128 vbi = *reinterpret_cast<const F128*>(bi + n);
  F128 vbf = *reinterpret_cast<const F128*>(bfv + n);
  F128 vbg = *reinterpret_cast<const F128*>(bg + n);
  F128 vbc = *reinterpret_cast<const F128*>(bc + n);
  F128 po  = *reinterpret_cast<const F128*>(prevout + idx);

  F128 cc, ss;
  float* zip = &zi.x; float* zfp = &zf.x; float* zgp = &zg.x; float* zcp = &zc.x;
  float* bip = &vbi.x; float* bfp = &vbf.x; float* bgp = &vbg.x; float* bcp = &vbc.x;
  float* pop = &po.x;  float* ccp = &cc.x; float* ssp = &ss.x;
  #pragma unroll
  for (int e = 0; e < 4; ++e) {
    float ig = 1.0f / (1.0f + __expf(-(zip[e] + bip[e])));
    float fg = 1.0f / (1.0f + __expf(-(zfp[e] + bfp[e])));
    float gg = 1.0f / (1.0f + __expf(-(zgp[e] + bgp[e])));
    float c_ = tanhf(zcp[e] + bcp[e]);
    float c  = fg * pop[e] + ig * c_;
    ccp[e] = c;
    ssp[e] = gg * tanhf(c);
  }
  *reinterpret_cast<F128*>(out + idx)            = cc;  // c
  *reinterpret_cast<F128*>(out + BH + idx)       = ss;  // state
  *reinterpret_cast<F128*>(out + 2 * BH + idx)   = cc;  // c (stacked)
}

// ============================================================
extern "C" void kernel_launch(void* const* d_in, const int* in_sizes, int n_in,
                              void* d_out, int out_size, void* d_ws, size_t ws_size,
                              hipStream_t stream) {
  const float* inputs = (const float*)d_in[0];
  const float* states = (const float*)d_in[1];
  const float* Wi = (const float*)d_in[2];
  const float* Ui = (const float*)d_in[3];
  const float* bi = (const float*)d_in[4];
  const float* Wf = (const float*)d_in[5];
  const float* Uf = (const float*)d_in[6];
  const float* bfv = (const float*)d_in[7];
  const float* Wg = (const float*)d_in[8];
  const float* Ug = (const float*)d_in[9];
  const float* bg = (const float*)d_in[10];
  const float* Wc = (const float*)d_in[11];
  const float* Uc = (const float*)d_in[12];
  const float* bc = (const float*)d_in[13];
  float* out = (float*)d_out;

  const float* st0 = states;                       // prevstate
  const float* st1 = states + (size_t)TB * TH;     // prevoutput

  unsigned short* Xbf = (unsigned short*)d_ws;                     // 4 MB
  float* Z = (float*)((char*)d_ws + (size_t)TB * TK * 2);          // 16 MB

  lstm_pack_x<<<(TB * TK) / 4 / 256, 256, 0, stream>>>(inputs, st0, Xbf);

  dim3 gg(TH / BN, TB / BM, 4);
  lstm_gemm4<<<gg, 256, 0, stream>>>(Xbf, Wi, Ui, Wf, Uf, Wg, Ug, Wc, Uc, Z);

  lstm_eltwise<<<((size_t)TB * TH) / 4 / 256, 256, 0, stream>>>(
      Z, bi, bfv, bg, bc, st1, out);
}
